// CustomSelfAttention_87247965651379
// MI455X (gfx1250) — matrix operand order repaired and checked by reading, not compiled, not run on hardware
//
#include <hip/hip_runtime.h>
#include <hip/hip_bf16.h>

typedef _Float16 half_t;
typedef __attribute__((ext_vector_type(16))) _Float16 v16h;
typedef __attribute__((ext_vector_type(8)))  float    v8f;
typedef __attribute__((ext_vector_type(4)))  float    f32x4;

#define SEQ   4096
#define EMB   768
#define NH    12
#define HD    64
#define NIDX  10   /* K_PAIRS * 2 */

// ---------------------------------------------------------------- converts
__global__ __launch_bounds__(256) void cvt_f32_f16(const float* __restrict__ s,
                                                   half_t* __restrict__ d, int n) {
  int i = blockIdx.x * blockDim.x + threadIdx.x;
  if (i < n) d[i] = (half_t)s[i];
}

// ---------------------------------------------------------------- GEMM: out = A @ W^T + bias
// A: MxK f16 row-major. W: NoutxK f16 row-major (so B[k][n] = W[n][k], K-contiguous per column).
// mode 0: outH token-major (MxNout f16)         [Q, K projections]
// mode 1: outH transposed  (Nout x M f16)       [V projection -> Vt[dim][token], b128 stores]
// mode 2: outF token-major (MxNout f32)         [final projection -> d_out]
__global__ __launch_bounds__(128) void gemm_xwT(const half_t* __restrict__ A,
                                                const half_t* __restrict__ W,
                                                const float* __restrict__ bias,
                                                half_t* __restrict__ outH,
                                                float* __restrict__ outF,
                                                int M, int Nout, int K, int mode)
{
  const int lane = threadIdx.x & 31;
  const int wv   = threadIdx.x >> 5;
  const int l16  = lane & 15, hi = lane >> 4;
  const int tilesN = Nout >> 4;
  const int tile = blockIdx.x * 4 + wv;
  const int mT = tile / tilesN;
  const int nT = tile - mT * tilesN;

  // A-frag: row = mT*16+l16, k elems kb..kb+15, kb = 16*hi
  const half_t* pa = A + (size_t)(mT * 16 + l16) * K + hi * 16;
  // B-frag: col = nT*16+l16, k elems kb..kb+15, kb = 16*hi (K-contiguous in W row)
  const half_t* pb = W + (size_t)(nT * 16 + l16) * K + hi * 16;

  v8f c = {};
  for (int k0 = 0; k0 < K; k0 += 32) {
    v16h a, b;
    ((f32x4*)&a)[0] = *(const f32x4*)(pa + k0);
    ((f32x4*)&a)[1] = *(const f32x4*)(pa + k0 + 8);
    ((f32x4*)&b)[0] = *(const f32x4*)(pb + k0);
    ((f32x4*)&b)[1] = *(const f32x4*)(pb + k0 + 8);
    c = __builtin_amdgcn_wmma_f32_16x16x32_f16(false, a, false, b, (short)0, c,
                                               false, false);
  }

  const int colG = nT * 16 + l16;
  const float bb = bias[colG];
  if (mode == 0) {
#pragma unroll
    for (int r = 0; r < 8; ++r)
      outH[(size_t)(mT * 16 + 8 * hi + r) * Nout + colG] = (half_t)(c[r] + bb);
  } else if (mode == 1) {
    __align__(16) half_t h8[8];
#pragma unroll
    for (int r = 0; r < 8; ++r) h8[r] = (half_t)(c[r] + bb);
    // D-frag rows (tokens) are consecutive per lane -> one b128 store
    *(f32x4*)(outH + (size_t)colG * M + mT * 16 + 8 * hi) = *(const f32x4*)h8;
  } else {
#pragma unroll
    for (int r = 0; r < 8; ++r)
      outF[(size_t)(mT * 16 + 8 * hi + r) * Nout + colG] = c[r] + bb;
  }
}

// ---------------------------------------------------------------- flash attention
// Q,Kx: [SEQ][EMB] f16 token-major. Vt: [EMB(dim-major per head)][SEQ] f16.
// sim: [SEQ][NIDX] int32.  AO: [SEQ][EMB] f16 attention output (pre-projection).
__global__ __launch_bounds__(128) void flash_attn(const half_t* __restrict__ Q,
                                                  const half_t* __restrict__ Kx,
                                                  const half_t* __restrict__ Vt,
                                                  const int* __restrict__ sim,
                                                  half_t* __restrict__ AO)
{
  __shared__ __align__(16) half_t Pbuf[4][16 * 32];
  __shared__ float Msh[4][16];

  const int lane = threadIdx.x & 31;
  const int wv   = threadIdx.x >> 5;
  const int l16  = lane & 15, hi = lane >> 4;
  const int gw = blockIdx.x * 4 + wv;
  const int h  = gw >> 8;      // 256 row-tiles per head
  const int rt = gw & 255;
  const int tok = rt * 16 + l16;

  // Q A-frags for K-dim 0..31 and 32..63 (kept live whole tile)
  const half_t* qp = Q + (size_t)tok * EMB + h * HD + hi * 16;
  v16h qa0, qa1;
  ((f32x4*)&qa0)[0] = *(const f32x4*)(qp + 0);
  ((f32x4*)&qa0)[1] = *(const f32x4*)(qp + 8);
  ((f32x4*)&qa1)[0] = *(const f32x4*)(qp + 32);
  ((f32x4*)&qa1)[1] = *(const f32x4*)(qp + 40);

  float Mr[8], Lr[8];
  v8f o0 = {}, o1 = {}, o2 = {}, o3 = {};
#pragma unroll
  for (int r = 0; r < 8; ++r) { Mr[r] = -1e30f; Lr[r] = 0.f; }

  for (int cb = 0; cb < SEQ; cb += 32) {
    // ---- S = (Q @ K^T) for 32 key columns (two 16-wide sub-tiles)
    const half_t* kp0 = Kx + (size_t)(cb + l16) * EMB + h * HD + hi * 16;
    const half_t* kp1 = kp0 + (size_t)16 * EMB;
    v16h b;
    v8f s0 = {}, s1 = {};
    ((f32x4*)&b)[0] = *(const f32x4*)(kp0 + 0);
    ((f32x4*)&b)[1] = *(const f32x4*)(kp0 + 8);
    s0 = __builtin_amdgcn_wmma_f32_16x16x32_f16(false, qa0, false, b, (short)0, s0, false, false);
    ((f32x4*)&b)[0] = *(const f32x4*)(kp0 + 32);
    ((f32x4*)&b)[1] = *(const f32x4*)(kp0 + 40);
    s0 = __builtin_amdgcn_wmma_f32_16x16x32_f16(false, qa1, false, b, (short)0, s0, false, false);
    ((f32x4*)&b)[0] = *(const f32x4*)(kp1 + 0);
    ((f32x4*)&b)[1] = *(const f32x4*)(kp1 + 8);
    s1 = __builtin_amdgcn_wmma_f32_16x16x32_f16(false, qa0, false, b, (short)0, s1, false, false);
    ((f32x4*)&b)[0] = *(const f32x4*)(kp1 + 32);
    ((f32x4*)&b)[1] = *(const f32x4*)(kp1 + 40);
    s1 = __builtin_amdgcn_wmma_f32_16x16x32_f16(false, qa1, false, b, (short)0, s1, false, false);

    // ---- online softmax update (row-wise reduce over 16-lane groups)
#pragma unroll
    for (int r = 0; r < 8; ++r) {
      float a0 = s0[r] * 0.125f, a1 = s1[r] * 0.125f;   // d^-0.5 = 1/8
      float t = fmaxf(a0, a1);
      t = fmaxf(t, __shfl_xor(t, 1, 32));
      t = fmaxf(t, __shfl_xor(t, 2, 32));
      t = fmaxf(t, __shfl_xor(t, 4, 32));
      t = fmaxf(t, __shfl_xor(t, 8, 32));
      float nM = fmaxf(Mr[r], t);
      float al = __expf(Mr[r] - nM);
      Mr[r] = nM;
      float p0 = __expf(a0 - nM);
      float p1 = __expf(a1 - nM);
      float rs = p0 + p1;
      rs += __shfl_xor(rs, 1, 32);
      rs += __shfl_xor(rs, 2, 32);
      rs += __shfl_xor(rs, 4, 32);
      rs += __shfl_xor(rs, 8, 32);
      Lr[r] = Lr[r] * al + rs;
      o0[r] *= al; o1[r] *= al; o2[r] *= al; o3[r] *= al;
      int m = 8 * hi + r;
      Pbuf[wv][m * 32 + l16]      = (half_t)p0;
      Pbuf[wv][m * 32 + l16 + 16] = (half_t)p1;
    }

    // ---- P (16x32) as A-frag via per-wave LDS bounce (C-layout -> A-layout transpose)
    v16h pa;
    ((f32x4*)&pa)[0] = *(const f32x4*)&Pbuf[wv][l16 * 32 + hi * 16];
    ((f32x4*)&pa)[1] = *(const f32x4*)&Pbuf[wv][l16 * 32 + hi * 16 + 8];

    // ---- O += P @ V  (V B-frags contiguous because Vt is dim-major)
    const half_t* vp = Vt + (size_t)(h * HD + l16) * SEQ + cb + hi * 16;
    v16h vb;
    ((f32x4*)&vb)[0] = *(const f32x4*)(vp + 0);
    ((f32x4*)&vb)[1] = *(const f32x4*)(vp + 8);
    o0 = __builtin_amdgcn_wmma_f32_16x16x32_f16(false, pa, false, vb, (short)0, o0, false, false);
    vp += (size_t)16 * SEQ;
    ((f32x4*)&vb)[0] = *(const f32x4*)(vp + 0);
    ((f32x4*)&vb)[1] = *(const f32x4*)(vp + 8);
    o1 = __builtin_amdgcn_wmma_f32_16x16x32_f16(false, pa, false, vb, (short)0, o1, false, false);
    vp += (size_t)16 * SEQ;
    ((f32x4*)&vb)[0] = *(const f32x4*)(vp + 0);
    ((f32x4*)&vb)[1] = *(const f32x4*)(vp + 8);
    o2 = __builtin_amdgcn_wmma_f32_16x16x32_f16(false, pa, false, vb, (short)0, o2, false, false);
    vp += (size_t)16 * SEQ;
    ((f32x4*)&vb)[0] = *(const f32x4*)(vp + 0);
    ((f32x4*)&vb)[1] = *(const f32x4*)(vp + 8);
    o3 = __builtin_amdgcn_wmma_f32_16x16x32_f16(false, pa, false, vb, (short)0, o3, false, false);
  }

  // ---- exact sparse bias correction: exp(s+b) = exp(s) + (e^b - 1) exp(s)
  if (l16 == 0) {
#pragma unroll
    for (int r = 0; r < 8; ++r) Msh[wv][8 * hi + r] = Mr[r];
  }
  const int* simr = sim + (size_t)rt * 16 * NIDX;
#pragma unroll
  for (int m = 0; m < 16; ++m) {
    const int* ip = simr + m * NIDX;
    const float Mrow = Msh[wv][m];
    const half_t* qrow = Q + (size_t)(rt * 16 + m) * EMB + h * HD;
    float extra = 0.f;
    for (int p = 0; p < NIDX; ++p) {
      int j = ip[p];
      if ((unsigned)j >= SEQ) continue;              // matches reference idx<n gating
      int mult = 0, first = 1;
      for (int q = 0; q < NIDX; ++q) {               // duplicate indices accumulate
        int jq = ip[q];
        mult += (jq == j);
        if (q < p && jq == j) first = 0;
      }
      if (!first) continue;                          // uniform branch (same idx all lanes)
      const half_t* krow = Kx + (size_t)j * EMB + h * HD;
      float dacc = (float)qrow[2 * lane]     * (float)krow[2 * lane]
                 + (float)qrow[2 * lane + 1] * (float)krow[2 * lane + 1];
      dacc += __shfl_xor(dacc, 1, 32);
      dacc += __shfl_xor(dacc, 2, 32);
      dacc += __shfl_xor(dacc, 4, 32);
      dacc += __shfl_xor(dacc, 8, 32);
      dacc += __shfl_xor(dacc, 16, 32);
      float w = (__expf((float)mult) - 1.f) * __expf(dacc * 0.125f - Mrow);
      extra += w;
      if (hi == (m >> 3)) {                          // lanes owning row m in C-layout
        const size_t vco = (size_t)(h * HD + l16) * SEQ + (size_t)j;
        o0[m & 7] += w * (float)Vt[vco];
        o1[m & 7] += w * (float)Vt[vco + (size_t)16 * SEQ];
        o2[m & 7] += w * (float)Vt[vco + (size_t)32 * SEQ];
        o3[m & 7] += w * (float)Vt[vco + (size_t)48 * SEQ];
      }
    }
    if (hi == (m >> 3)) Lr[m & 7] += extra;
  }

  // ---- normalize and store
#pragma unroll
  for (int r = 0; r < 8; ++r) {
    float inv = 1.f / Lr[r];
    size_t base = (size_t)(rt * 16 + 8 * hi + r) * EMB + h * HD + l16;
    AO[base]      = (half_t)(o0[r] * inv);
    AO[base + 16] = (half_t)(o1[r] * inv);
    AO[base + 32] = (half_t)(o2[r] * inv);
    AO[base + 48] = (half_t)(o3[r] * inv);
  }
}

// ---------------------------------------------------------------- launch
extern "C" void kernel_launch(void* const* d_in, const int* in_sizes, int n_in,
                              void* d_out, int out_size, void* d_ws, size_t ws_size,
                              hipStream_t stream)
{
  (void)in_sizes; (void)n_in; (void)out_size; (void)ws_size;
  const float* x  = (const float*)d_in[0];
  const int*  sim = (const int*)d_in[1];
  const float* Wq = (const float*)d_in[2];
  const float* bq = (const float*)d_in[3];
  const float* Wk = (const float*)d_in[4];
  const float* bk = (const float*)d_in[5];
  const float* Wv = (const float*)d_in[6];
  const float* bv = (const float*)d_in[7];
  const float* Wo = (const float*)d_in[8];
  const float* bo = (const float*)d_in[9];
  float* out = (float*)d_out;

  const size_t NC = (size_t)SEQ * EMB;   // 3,145,728 halves
  const size_t CC = (size_t)EMB * EMB;   //   589,824 halves
  half_t* ws  = (half_t*)d_ws;
  half_t* xh  = ws;
  half_t* wqh = xh + NC;
  half_t* wkh = wqh + CC;
  half_t* wvh = wkh + CC;
  half_t* woh = wvh + CC;
  half_t* qh  = woh + CC;
  half_t* kh  = qh + NC;
  half_t* vt  = kh + NC;   // [EMB][SEQ] dim-major V
  half_t* ao  = vt + NC;

  const int nx = (int)NC, nw = (int)CC;
  cvt_f32_f16<<<(nx + 255) / 256, 256, 0, stream>>>(x,  xh,  nx);
  cvt_f32_f16<<<(nw + 255) / 256, 256, 0, stream>>>(Wq, wqh, nw);
  cvt_f32_f16<<<(nw + 255) / 256, 256, 0, stream>>>(Wk, wkh, nw);
  cvt_f32_f16<<<(nw + 255) / 256, 256, 0, stream>>>(Wv, wvh, nw);
  cvt_f32_f16<<<(nw + 255) / 256, 256, 0, stream>>>(Wo, woh, nw);

  const int gemmBlocks = (SEQ / 16) * (EMB / 16) / 4;   // 3072 blocks x 4 waves
  gemm_xwT<<<gemmBlocks, 128, 0, stream>>>(xh, wqh, bq, qh, nullptr, SEQ, EMB, EMB, 0);
  gemm_xwT<<<gemmBlocks, 128, 0, stream>>>(xh, wkh, bk, kh, nullptr, SEQ, EMB, EMB, 0);
  gemm_xwT<<<gemmBlocks, 128, 0, stream>>>(xh, wvh, bv, vt, nullptr, SEQ, EMB, EMB, 1);

  flash_attn<<<NH * (SEQ / 16) / 4, 128, 0, stream>>>(qh, kh, vt, sim, ao);

  gemm_xwT<<<gemmBlocks, 128, 0, stream>>>(ao, woh, bo, nullptr, out, SEQ, EMB, EMB, 2);
}
